// RGB_SFMCNN_43911745635156
// MI455X (gfx1250) — matrix operand order repaired and checked by reading, not compiled, not run on hardware
//
#include <hip/hip_runtime.h>
#include <hip/hip_bf16.h>
#include <math.h>

typedef __attribute__((ext_vector_type(16))) _Float16 v16h;
typedef __attribute__((ext_vector_type(8)))  _Float16 v8h;
typedef __attribute__((ext_vector_type(8)))  float    v8f;

#define NB 32  // batch

// ---------------- grayscale ----------------
__global__ void gray_k(const float* __restrict__ x, float* __restrict__ g) {
    int i = blockIdx.x * blockDim.x + threadIdx.x;
    const int total = NB * 10000;
    if (i >= total) return;
    int b = i / 10000, p = i - b * 10000;
    const float* xb = x + (long)b * 3 * 10000;
    g[i] = 0.2989f * xb[p] + 0.587f * xb[10000 + p] + 0.114f * xb[20000 + p];
}

// ---------------- rgb RBF conv (75 out ch, k=5, no pad) ----------------
__global__ void rgb_conv_k(const float* __restrict__ x, const float* __restrict__ rw,
                           float* __restrict__ outcat) {
    int idx = blockIdx.x * blockDim.x + threadIdx.x;
    const int total = NB * 9216;
    if (idx >= total) return;
    int b = idx / 9216, l = idx - b * 9216;
    int y = l / 96, xx = l - y * 96;
    float s1[3] = {0.f, 0.f, 0.f};
    float s2 = 0.f;
    for (int c = 0; c < 3; c++) {
        const float* base = x + (((long)b * 3 + c) * 100 + y) * 100 + xx;
        for (int dy = 0; dy < 5; dy++)
            for (int dx = 0; dx < 5; dx++) {
                float v = base[dy * 100 + dx];
                s1[c] += v; s2 += v * v;
            }
    }
    for (int o = 0; o < 75; o++) {
        float dot = 0.f, ww = 0.f;
        for (int c = 0; c < 3; c++) { float w = rw[o * 3 + c]; dot += w * s1[c]; ww += w * w; }
        float d2 = s2 + 25.f * ww - 2.f * dot;
        outcat[(((long)b * 100) + o) * 9216 + l] = sqrtf(fmaxf(d2, 0.f));
    }
}

// ---------------- gray RBF conv (25 out ch, k=5, no pad) ----------------
__global__ void gray_conv_k(const float* __restrict__ g, const float* __restrict__ gw,
                            float* __restrict__ outcat) {
    int idx = blockIdx.x * blockDim.x + threadIdx.x;
    const int total = NB * 9216;
    if (idx >= total) return;
    int b = idx / 9216, l = idx - b * 9216;
    int y = l / 96, xx = l - y * 96;
    float p[25];
    float s2 = 0.f;
    const float* base = g + ((long)b * 100 + y) * 100 + xx;
    for (int dy = 0; dy < 5; dy++)
        for (int dx = 0; dx < 5; dx++) {
            float v = base[dy * 100 + dx];
            p[dy * 5 + dx] = v; s2 += v * v;
        }
    for (int o = 0; o < 25; o++) {
        float dot = 0.f, ww = 0.f;
        for (int k = 0; k < 25; k++) { float w = gw[o * 25 + k]; dot += w * p[k]; ww += w * w; }
        float d2 = s2 + ww - 2.f * dot;
        outcat[(((long)b * 100) + 75 + o) * 9216 + l] = sqrtf(fmaxf(d2, 0.f));
    }
}

// ---------------- weighted 2x2 pool (sfm) ----------------
__global__ void sfm_k(const float* __restrict__ in, float* __restrict__ out,
                      int C, int oh, int ow) {
    long idx = (long)blockIdx.x * blockDim.x + threadIdx.x;
    long total = (long)NB * C * oh * ow;
    if (idx >= total) return;
    int j = idx % ow; long t = idx / ow;
    int i = t % oh;  t /= oh;
    int c = t % C;   int b = t / C;
    int iw = ow * 2;
    const float* p = in + ((((long)b * C + c) * (oh * 2)) + 2 * i) * iw + 2 * j;
    out[idx] = (p[0] * 0.90f + p[1] * 0.93f + p[iw] * 0.96f + p[iw + 1] * 0.99f) * 0.25f;
}

// ---------------- sum over channels of x^2 ----------------
__global__ void sumsq_ch_k(const float* __restrict__ in, float* __restrict__ out,
                           int C, int HW) {
    int idx = blockIdx.x * blockDim.x + threadIdx.x;
    int total = NB * HW;
    if (idx >= total) return;
    int b = idx / HW, p = idx - b * HW;
    float s = 0.f;
    for (int c = 0; c < C; c++) { float v = in[((long)b * C + c) * HW + p]; s += v * v; }
    out[idx] = s;
}

// ---------------- 3x3 box sums ----------------
__global__ void box3_pad_k(const float* __restrict__ t, float* __restrict__ out, int H, int W) {
    int idx = blockIdx.x * blockDim.x + threadIdx.x;
    int total = NB * H * W;
    if (idx >= total) return;
    int b = idx / (H * W), l = idx - b * (H * W);
    int y = l / W, x = l - y * W;
    float s = 0.f;
    for (int dy = -1; dy <= 1; dy++)
        for (int dx = -1; dx <= 1; dx++) {
            int iy = y + dy, ix = x + dx;
            if (iy >= 0 && iy < H && ix >= 0 && ix < W) s += t[(long)b * H * W + iy * W + ix];
        }
    out[idx] = s;
}

__global__ void box3_nopad_k(const float* __restrict__ t, float* __restrict__ out, int H, int W) {
    int OH = H - 2, OW = W - 2;
    int idx = blockIdx.x * blockDim.x + threadIdx.x;
    int total = NB * OH * OW;
    if (idx >= total) return;
    int b = idx / (OH * OW), l = idx - b * (OH * OW);
    int y = l / OW, x = l - y * OW;
    float s = 0.f;
    for (int dy = 0; dy < 3; dy++)
        for (int dx = 0; dx < 3; dx++) s += t[(long)b * H * W + (y + dy) * W + (x + dx)];
    out[idx] = s;
}

// ---------------- row-wise sum of squares of weights ----------------
__global__ void rowsq_k(const float* __restrict__ w, float* __restrict__ out, int M, int K) {
    int o = blockIdx.x * blockDim.x + threadIdx.x;
    if (o >= M) return;
    float s = 0.f;
    for (int k = 0; k < K; k++) { float v = w[(long)o * K + k]; s += v * v; }
    out[o] = s;
}

// ---------------- weights f32 -> padded f16 (Mpad x Kpad, zero-filled) ----------------
__global__ void cvt_w_k(const float* __restrict__ w, _Float16* __restrict__ out,
                        int M, int K, int Kpad, int total) {
    int idx = blockIdx.x * blockDim.x + threadIdx.x;
    if (idx >= total) return;
    int m = idx / Kpad, k = idx - m * Kpad;
    float v = (m < M && k < K) ? w[(long)m * K + k] : 0.f;
    out[idx] = (_Float16)v;
}

// ---------------- im2col -> padded f16, n-major with K contiguous ----------------
// out[b][n][k], n in [0,Npad), k in [0,Kpad); zero-filled padding.
__global__ void im2col_f16_k(const float* __restrict__ in, _Float16* __restrict__ out,
                             int Cin, int IH, int IW, int OH, int OW, int pad,
                             int Ksz, int Kpad, int Npad) {
    long idx = (long)blockIdx.x * blockDim.x + threadIdx.x;
    long total = (long)NB * Npad * Kpad;
    if (idx >= total) return;
    int k = idx % Kpad; long t = idx / Kpad;
    int n = t % Npad;   int b = t / Npad;
    int N = OH * OW;
    float v = 0.f;
    if (k < Ksz && n < N) {
        int c = k / 9;
        int r = k - c * 9;
        int dy = r / 3;
        int dx = r - dy * 3;
        int y = n / OW, x = n - (n / OW) * OW;
        int iy = y + dy - pad, ix = x + dx - pad;
        if (iy >= 0 && iy < IH && ix >= 0 && ix < IW)
            v = in[(((long)b * Cin + c) * IH + iy) * IW + ix];
    }
    out[idx] = (_Float16)v;
}

// ---------------- RBF distance GEMM on padded f16 operands ----------------
// d[b,m,n] = sqrt(max(0, ||P_n||^2 + ||w_m||^2 - 2 * (w_m . P_n)))
// One wave per 16x16 tile, K stepped by 32 with v_wmma_f32_16x16x32_f16.
// All loads unconditional (operands fully padded); only stores guarded.
__global__ void rbf_gemm_k(const _Float16* __restrict__ A,   // Mpad x Kpad
                           const _Float16* __restrict__ Bm,  // NB x Npad x Kpad
                           const float* __restrict__ pn2, const float* __restrict__ wo2,
                           float* __restrict__ out,
                           int M, int N, int Npad, int Kpad) {
    int lane = threadIdx.x & 31;
    int wave = threadIdx.x >> 5;
    int n0 = (blockIdx.x * (blockDim.x >> 5) + wave) * 16;
    int m0 = blockIdx.y * 16;
    int b  = blockIdx.z;

    int row  = m0 + (lane & 15);          // A row (filter m)
    int klo  = (lane >> 4) * 8;           // A half-lane K offset
    int coln = n0 + (lane & 15);          // column (output pixel n)
    int kb16 = (lane >> 4) * 16;          // unused; B loads use klo, same k order as A

    const _Float16* aptr = A + (long)row * Kpad;
    const _Float16* bptr = Bm + ((long)b * Npad + coln) * Kpad;

    v8f acc = {};
    for (int k0 = 0; k0 < Kpad; k0 += 32) {
        v8h alo = *(const v8h*)(aptr + k0 + klo);
        v8h ahi = *(const v8h*)(aptr + k0 + 16 + klo);
        v16h av = __builtin_shufflevector(alo, ahi,
                                          0, 1, 2, 3, 4, 5, 6, 7,
                                          8, 9, 10, 11, 12, 13, 14, 15);
        v8h blo = *(const v8h*)(bptr + k0 + klo);
        v8h bhi = *(const v8h*)(bptr + k0 + 16 + klo);
        v16h bv = __builtin_shufflevector(blo, bhi,
                                          0, 1, 2, 3, 4, 5, 6, 7,
                                          8, 9, 10, 11, 12, 13, 14, 15);
        __builtin_prefetch((const void*)((const char*)(bptr + k0) + 256), 0, 1);
        acc = __builtin_amdgcn_wmma_f32_16x16x32_f16(false, av, false, bv,
                                                     (short)0, acc, false, false);
    }
    if (coln >= N) return;
    float p2 = pn2[(long)b * N + coln];
#pragma unroll
    for (int v = 0; v < 8; v++) {
        int m = m0 + ((lane < 16) ? v : 8 + v);
        if (m < M) {
            float d2 = p2 + wo2[m] - 2.f * acc[v];
            out[(((long)b * M) + m) * N + coln] = sqrtf(fmaxf(d2, 0.f));
        }
    }
}

// ---------------- triangle: 2-level histogram k-th order statistic ----------------
__global__ void fill_u32_k(unsigned* p, int n) {
    int i = blockIdx.x * blockDim.x + threadIdx.x;
    if (i < n) p[i] = 0u;
}

__global__ void init_sel_k(float* lo, float* hi, long long* kkarr, const float* wsc,
                           long long kk) {
    int b = threadIdx.x;
    if (b < NB) { lo[b] = 0.f; hi[b] = wsc[0]; kkarr[b] = kk; }
}

__global__ void hist_k(const float* __restrict__ d, long bstride, long off, long len,
                       const float* __restrict__ lo, const float* __restrict__ hi,
                       unsigned* __restrict__ hist) {
    int b = blockIdx.y;
    const float* p = d + (long)b * bstride + off;
    float l = lo[b], h = hi[b];
    float inv = 1024.f / (h - l);
    long stride = (long)gridDim.x * blockDim.x;
    for (long i = (long)blockIdx.x * blockDim.x + threadIdx.x; i < len; i += stride) {
        float v = p[i];
        if (v >= l && v < h) {
            int bin = (int)((v - l) * inv);
            if (bin > 1023) bin = 1023;
            atomicAdd(&hist[b * 1024 + bin], 1u);
        }
    }
}

__global__ void select_k(const unsigned* __restrict__ hist, float* lo, float* hi,
                         long long* kkarr, const float* wsc, float* thr, int final_pass) {
    int b = threadIdx.x;
    if (b >= NB) return;
    long long kk = kkarr[b];
    float l = lo[b], h = hi[b];
    float width = (h - l) * (1.f / 1024.f);
    long long cum = 0;
    int bin = 1023;
    for (int i = 0; i < 1024; i++) {
        unsigned c = hist[b * 1024 + i];
        if (cum + (long long)c >= kk) { bin = i; break; }
        cum += c;
    }
    if (final_pass) {
        thr[b] = fminf(l + (bin + 1) * width, wsc[0]);
    } else {
        lo[b] = l + bin * width;
        hi[b] = l + (bin + 1) * width;
        kkarr[b] = kk - cum;
    }
}

__global__ void tri_apply_k(float* __restrict__ d, long bstride, long off, long len,
                            const float* __restrict__ thr, const float* __restrict__ wsc) {
    int b = blockIdx.y;
    float w = wsc[0];
    float t = thr[b];
    float invw = 1.f / w;
    float* p = d + (long)b * bstride + off;
    long stride = (long)gridDim.x * blockDim.x;
    for (long i = (long)blockIdx.x * blockDim.x + threadIdx.x; i < len; i += stride) {
        float v = p[i];
        if (v > t) v = w;
        p[i] = 1.f - v * invw;
    }
}

// ---------------- FC ----------------
__global__ void fc_k(const float* __restrict__ act, const float* __restrict__ w,
                     const float* __restrict__ bias, float* __restrict__ out) {
    int o = blockIdx.x, b = blockIdx.y;
    __shared__ float red[256];
    const float* a  = act + (long)b * 302500;
    const float* wr = w + (long)o * 302500;
    float s = 0.f;
    for (int i = threadIdx.x; i < 302500; i += 256) s += a[i] * wr[i];
    red[threadIdx.x] = s;
    __syncthreads();
    for (int offv = 128; offv > 0; offv >>= 1) {
        if (threadIdx.x < offv) red[threadIdx.x] += red[threadIdx.x + offv];
        __syncthreads();
    }
    if (threadIdx.x == 0) out[b * 10 + o] = red[0] + bias[o];
}

// ---------------- driver ----------------
extern "C" void kernel_launch(void* const* d_in, const int* in_sizes, int n_in,
                              void* d_out, int out_size, void* d_ws, size_t ws_size,
                              hipStream_t stream) {
    const float* x       = (const float*)d_in[0];
    const float* rgb_w   = (const float*)d_in[1];
    const float* gray_w  = (const float*)d_in[2];
    const float* conv1_w = (const float*)d_in[3];
    const float* conv2_w = (const float*)d_in[4];
    const float* fc_w    = (const float*)d_in[5];
    const float* fc_b    = (const float*)d_in[6];
    const float* w1      = (const float*)d_in[7];
    const float* w2      = (const float*)d_in[8];
    const float* w3      = (const float*)d_in[9];
    float* out = (float*)d_out;

    char* base = (char*)d_ws;
    size_t off = 0;
    auto carve = [&](size_t bytes) -> void* {
        void* p = (void*)(base + off);
        off += (bytes + 255) & ~(size_t)255;
        return p;
    };

    // persistent small buffers
    float*     tmpa  = (float*)carve((size_t)NB * 2304 * 4);
    float*     pn2a  = (float*)carve((size_t)NB * 2304 * 4);
    float*     tmpb  = (float*)carve((size_t)NB * 576 * 4);
    float*     pn2b  = (float*)carve((size_t)NB * 484 * 4);
    float*     wo2a  = (float*)carve(240 * 4);
    float*     wo2b  = (float*)carve(640 * 4);
    unsigned*  hist  = (unsigned*)carve((size_t)NB * 1024 * 4);
    float*     lo    = (float*)carve(NB * 4);
    float*     hi    = (float*)carve(NB * 4);
    float*     thr   = (float*)carve(NB * 4);
    long long* kkarr = (long long*)carve(NB * 8);
    _Float16*  wA1   = (_Float16*)carve((size_t)240 * 928 * 2);
    _Float16*  wA2   = (_Float16*)carve((size_t)640 * 2048 * 2);

    // stage-lifetime buffers
    float* sfm1b = (float*)carve((size_t)NB * 100 * 2304 * 4);   // (B,100,48,48)
    float* dist1 = (float*)carve((size_t)NB * 225 * 2304 * 4);   // (B,225,48,48)
    float* act1  = (float*)carve((size_t)NB * 225 * 576 * 4);    // (B,225,24,24)
    float* dist2 = (float*)carve((size_t)NB * 625 * 484 * 4);    // (B,625,22,22)

    // big overlay region F: {gbuf+cat} (stages 1-3) -> patch1 (stage 4) -> patch2 (stage 6)
    size_t szPatch1 = (size_t)NB * 2304 * 928 * 2;   // ~137 MB
    char* regF = (char*)carve(szPatch1);
    float*    gbuf   = (float*)regF;                                   // 1.28 MB
    float*    cat    = (float*)(regF + (((size_t)NB * 10000 * 4 + 255) & ~(size_t)255)); // 118 MB
    _Float16* patch1 = (_Float16*)regF;                                // 2304x928 per batch
    _Float16* patch2 = (_Float16*)regF;                                // 512x2048 per batch

    auto triangle = [&](float* data, long bstride, long offe, long len, long long kk,
                        const float* wsc) {
        fill_u32_k<<<(NB * 1024 + 255) / 256, 256, 0, stream>>>(hist, NB * 1024);
        init_sel_k<<<1, NB, 0, stream>>>(lo, hi, kkarr, wsc, kk);
        hist_k<<<dim3(256, NB), 256, 0, stream>>>(data, bstride, offe, len, lo, hi, hist);
        select_k<<<1, NB, 0, stream>>>(hist, lo, hi, kkarr, wsc, thr, 0);
        fill_u32_k<<<(NB * 1024 + 255) / 256, 256, 0, stream>>>(hist, NB * 1024);
        hist_k<<<dim3(256, NB), 256, 0, stream>>>(data, bstride, offe, len, lo, hi, hist);
        select_k<<<1, NB, 0, stream>>>(hist, lo, hi, kkarr, wsc, thr, 1);
        tri_apply_k<<<dim3(128, NB), 256, 0, stream>>>(data, bstride, offe, len, thr, wsc);
    };

    // Stage 1: grayscale + first-layer RBF convs into concat buffer (B,100,96,96)
    gray_k<<<(NB * 10000 + 255) / 256, 256, 0, stream>>>(x, gbuf);
    rgb_conv_k<<<(NB * 9216 + 255) / 256, 256, 0, stream>>>(x, rgb_w, cat);
    gray_conv_k<<<(NB * 9216 + 255) / 256, 256, 0, stream>>>(gbuf, gray_w, cat);

    // Stage 2: per-region triangle (rgb: 75*9216, gray: 25*9216), percent 0.5
    triangle(cat, 100L * 9216, 0, 75L * 9216, 345600LL, w1);
    triangle(cat, 100L * 9216, 75L * 9216, 25L * 9216, 115200LL, w1);

    // Stage 3: sfm -> (B,100,48,48)
    sfm_k<<<(int)(((long)NB * 100 * 2304 + 255) / 256), 256, 0, stream>>>(cat, sfm1b, 100, 48, 48);

    // Stage 4: conv1 RBF distance, M=225 K=900(928) N=2304
    sumsq_ch_k<<<(NB * 2304 + 255) / 256, 256, 0, stream>>>(sfm1b, tmpa, 100, 2304);
    box3_pad_k<<<(NB * 2304 + 255) / 256, 256, 0, stream>>>(tmpa, pn2a, 48, 48);
    rowsq_k<<<1, 256, 0, stream>>>(conv1_w, wo2a, 225, 900);
    cvt_w_k<<<(240 * 928 + 255) / 256, 256, 0, stream>>>(conv1_w, wA1, 225, 900, 928, 240 * 928);
    {
        long tot = (long)NB * 2304 * 928;
        im2col_f16_k<<<(int)((tot + 255) / 256), 256, 0, stream>>>(sfm1b, patch1,
                                                                   100, 48, 48, 48, 48, 1,
                                                                   900, 928, 2304);
    }
    rbf_gemm_k<<<dim3(36, 15, NB), 128, 0, stream>>>(wA1, patch1, pn2a, wo2a, dist1,
                                                     225, 2304, 2304, 928);

    // Stage 5: triangle (percent 0.4 over 518400) then sfm -> (B,225,24,24)
    triangle(dist1, 225L * 2304, 0, 518400L, 207360LL, w2);
    sfm_k<<<(int)(((long)NB * 225 * 576 + 255) / 256), 256, 0, stream>>>(dist1, act1, 225, 24, 24);

    // Stage 6: conv2 RBF distance, M=625 K=2025(2048) N=484(512)
    sumsq_ch_k<<<(NB * 576 + 255) / 256, 256, 0, stream>>>(act1, tmpb, 225, 576);
    box3_nopad_k<<<(NB * 484 + 255) / 256, 256, 0, stream>>>(tmpb, pn2b, 24, 24);
    rowsq_k<<<3, 256, 0, stream>>>(conv2_w, wo2b, 625, 2025);
    cvt_w_k<<<(640 * 2048 + 255) / 256, 256, 0, stream>>>(conv2_w, wA2, 625, 2025, 2048,
                                                          640 * 2048);
    {
        long tot = (long)NB * 512 * 2048;
        im2col_f16_k<<<(int)((tot + 255) / 256), 256, 0, stream>>>(act1, patch2,
                                                                   225, 24, 24, 22, 22, 0,
                                                                   2025, 2048, 512);
    }
    rbf_gemm_k<<<dim3(8, 40, NB), 128, 0, stream>>>(wA2, patch2, pn2b, wo2b, dist2,
                                                    625, 484, 512, 2048);

    // Stage 7: triangle (percent 0.4 over 302500) in-place, then FC
    triangle(dist2, 302500L, 0, 302500L, 121000LL, w3);
    fc_k<<<dim3(10, NB), 256, 0, stream>>>(dist2, fc_w, fc_b, out);
}
